// ExtractionLayer_34471407518349
// MI455X (gfx1250) — matrix operand order repaired and checked by reading, not compiled, not run on hardware
//
#include <hip/hip_runtime.h>

// Problem constants (match reference)
#define B_DIM 32
#define V_DIM 256
#define F_DIM 128
#define P_DIM 64

typedef __attribute__((ext_vector_type(16))) _Float16 v16h;
typedef __attribute__((ext_vector_type(8)))  float    v8f;

// metric[b,v,f] = sum_p softmax(logits[b,f,:])[p] * exp(-0.5*((vol[v]*filt[f]-q2[b,p])*inv)^2)
//
// Per (b,f): a 16(v) x 32(p-chunk) Gaussian tile (f16) times a 32 x 16
// amplitude-broadcast tile (f16), f32-accumulated with v_wmma_f32_16x16x32_f16.
// exp(-0.5*(d*inv)^2) = exp2(-(cs*d)^2) with cs = inv*sqrt(0.5*log2(e)),
// so both q2 and vol*filt are pre-scaled by cs and the inner body is
// sub, mul(neg), v_exp_f32, cvt -> MACs ride the matrix pipe.
__global__ __launch_bounds__(512) void
extraction_metric_wmma(const float* __restrict__ q2,       // (B,P)
                       const float* __restrict__ logits,   // (B,1,F,P)
                       const float* __restrict__ volumes,  // (V,1)
                       const float* __restrict__ filters,  // (1,F)
                       const float* __restrict__ sigmap,   // scalar
                       float* __restrict__ out)            // (B,V,F)
{
    __shared__ float    s_ys[P_DIM];   // cs * q2[b,p]
    __shared__ float    s_e[P_DIM];    // exp(logit) scratch
    __shared__ _Float16 s_amp[P_DIM];  // softmax row, f16

    const int tid = threadIdx.x;
    const int bf  = blockIdx.x;          // 0 .. B*F-1
    const int b   = bf / F_DIM;
    const int f   = bf % F_DIM;

    const float sigma = sigmap[0];
    const float inv   = 1.0f / (sigma + 0.001f);
    const float cs    = inv * sqrtf(0.72134752044448170f);  // sqrt(0.5*log2(e))

    // ---- stage scaled q2 row + softmax(logits[b,f,:]) into LDS ----
    if (tid < P_DIM) {
        s_ys[tid] = cs * q2[b * P_DIM + tid];
        float l = logits[((size_t)b * F_DIM + f) * P_DIM + tid];
        s_e[tid] = __builtin_amdgcn_exp2f(l * 1.4426950408889634f);
    }
    __syncthreads();
    if (tid < P_DIM) {
        float s = 0.0f;
#pragma unroll
        for (int i = 0; i < P_DIM; ++i) s += s_e[i];   // broadcast LDS reads
        s_amp[tid] = (_Float16)(s_e[tid] / s);
    }
    __syncthreads();

    // ---- per-wave WMMA: wave w handles v-tile [16w, 16w+16) for this (b,f) ----
    const int wave = tid >> 5;        // 0..15
    const int lane = tid & 31;
    const int h    = lane >> 4;       // half-wave: 0 or 1
    const int m    = lane & 15;       // M (row within v-tile)
    const int v    = wave * 16 + m;

    const float xs = cs * volumes[v] * filters[f];

    v8f acc = {};
#pragma unroll
    for (int pc = 0; pc < 2; ++pc) {  // two K=32 chunks cover P=64
        const int pb = pc * 32;
        v16h A, Bm;
#pragma unroll
        for (int e = 0; e < 16; ++e) {
            // A-matrix 16x32 f16 layout: lanes<16 hold K in {0..7,16..23},
            // lanes>=16 hold K in {8..15,24..31}
            const int ka = (e < 8) ? (8 * h + e) : (8 + 8 * h + e);
            const float d = xs - s_ys[pb + ka];
            A[e] = (_Float16)__builtin_amdgcn_exp2f(-(d * d));
            // B-matrix 32x16 f16 layout: same K order as A,
            // value broadcast across all N columns
            Bm[e] = s_amp[pb + ka];
        }
        acc = __builtin_amdgcn_wmma_f32_16x16x32_f16(
            /*neg_a=*/false, A, /*neg_b=*/false, Bm,
            /*c_mod=*/(short)0, acc, /*reuse_a=*/false, /*reuse_b=*/false);
    }

    // D columns are identical; lane 0 holds M=0..7, lane 16 holds M=8..15.
    if (m == 0) {
        const int vb = wave * 16 + h * 8;
        float* o = out + ((size_t)b * V_DIM + vb) * F_DIM + f;
#pragma unroll
        for (int r = 0; r < 8; ++r) o[(size_t)r * F_DIM] = acc[r];
    }
}

extern "C" void kernel_launch(void* const* d_in, const int* in_sizes, int n_in,
                              void* d_out, int out_size, void* d_ws, size_t ws_size,
                              hipStream_t stream) {
    (void)in_sizes; (void)n_in; (void)d_ws; (void)ws_size; (void)out_size;
    const float* q2      = (const float*)d_in[0];   // (B,P)
    const float* logits  = (const float*)d_in[1];   // (B,1,F,P)
    const float* volumes = (const float*)d_in[2];   // (V,1)
    const float* filters = (const float*)d_in[3];   // (1,F)
    const float* sigma   = (const float*)d_in[4];   // scalar
    float* out = (float*)d_out;                     // (B,V,F)

    dim3 grid(B_DIM * F_DIM);   // 4096 blocks, one per (b,f)
    dim3 block(512);            // 16 waves; wave w -> v-tile w
    extraction_metric_wmma<<<grid, block, 0, stream>>>(q2, logits, volumes,
                                                       filters, sigma, out);
}
